// FP8Linear_52690658787834
// MI455X (gfx1250) — 8-bit matrix operand order repaired and checked by reading, compiled, not run on hardware
//
#include <hip/hip_runtime.h>
#include <hip/hip_bf16.h>
#include <stdint.h>

typedef __attribute__((ext_vector_type(16))) int      v16i;
typedef __attribute__((ext_vector_type(8)))  float    v8f;
typedef __attribute__((ext_vector_type(4)))  unsigned v4u;
typedef __attribute__((ext_vector_type(8)))  unsigned v8u;

#define FP8_MAX_V 448.0f
#define LDSPITCH  144
#define TILEBYTES (128 * LDSPITCH)
#define HALFBYTES (2 * TILEBYTES)

__global__ void init_amax_kernel(unsigned* amax_bits) {
    *amax_bits = __float_as_uint(1e-12f);
}

__global__ __launch_bounds__(256) void amax_kernel(const float* __restrict__ x,
                                                   unsigned* __restrict__ amax_bits,
                                                   int n4) {
    const float4* x4 = (const float4*)x;
    int i = blockIdx.x * blockDim.x + threadIdx.x;
    int stride = gridDim.x * blockDim.x;
    float m = 0.0f;
    for (int j = i; j < n4; j += stride) {
        float4 v = x4[j];
        m = fmaxf(m, fmaxf(fmaxf(fabsf(v.x), fabsf(v.y)),
                           fmaxf(fabsf(v.z), fabsf(v.w))));
    }
#pragma unroll
    for (int off = 16; off > 0; off >>= 1)
        m = fmaxf(m, __shfl_xor(m, off, 32));
    if ((threadIdx.x & 31) == 0)
        atomicMax(amax_bits, __float_as_uint(m));
}

__device__ __forceinline__ unsigned f2e4m3_sw(float f) {
    unsigned u = __float_as_uint(f);
    unsigned s = (u >> 24) & 0x80u;
    unsigned mag = u & 0x7FFFFFFFu;
    if (mag < 0x3C000000u) return s;
    if (mag > 0x43E00000u) mag = 0x43E00000u;
    unsigned e = (mag >> 23) - 127u + 7u;
    unsigned man = (mag >> 20) & 7u;
    return s | (e << 3) | man;
}

__device__ __forceinline__ unsigned pack4_fp8(float a, float b, float c, float d, float s) {
    a = fminf(fmaxf(a * s, -FP8_MAX_V), FP8_MAX_V);
    b = fminf(fmaxf(b * s, -FP8_MAX_V), FP8_MAX_V);
    c = fminf(fmaxf(c * s, -FP8_MAX_V), FP8_MAX_V);
    d = fminf(fmaxf(d * s, -FP8_MAX_V), FP8_MAX_V);
#if __has_builtin(__builtin_amdgcn_cvt_pk_fp8_f32)
    int v = __builtin_amdgcn_cvt_pk_fp8_f32(a, b, 0, false);
    v = __builtin_amdgcn_cvt_pk_fp8_f32(c, d, v, true);
    return (unsigned)v;
#else
    return f2e4m3_sw(a) | (f2e4m3_sw(b) << 8) | (f2e4m3_sw(c) << 16) | (f2e4m3_sw(d) << 24);
#endif
}

__global__ __launch_bounds__(256) void quantT_kernel(const float* __restrict__ x,
                                                     uint8_t* __restrict__ xqT,
                                                     const unsigned* __restrict__ amax_bits,
                                                     int M, int K) {
    __shared__ __align__(16) uint8_t tile[64 * 80];
    const float s = FP8_MAX_V / __uint_as_float(*amax_bits);
    const int t = threadIdx.x;
    const int mBase = blockIdx.x * 64;
    const int kBase = blockIdx.y * 64;
    {
        const int mr = t >> 2;
        const int kc = (t & 3) << 4;
        const float4* src = (const float4*)(x + (size_t)(mBase + mr) * K + kBase + kc);
        float4 f0 = src[0], f1 = src[1], f2 = src[2], f3 = src[3];
        uint4 o;
        o.x = pack4_fp8(f0.x, f0.y, f0.z, f0.w, s);
        o.y = pack4_fp8(f1.x, f1.y, f1.z, f1.w, s);
        o.z = pack4_fp8(f2.x, f2.y, f2.z, f2.w, s);
        o.w = pack4_fp8(f3.x, f3.y, f3.z, f3.w, s);
        *(uint4*)(tile + mr * 80 + kc) = o;
    }
    __syncthreads();
    {
        const int kr = t >> 2;
        const int mc = (t & 3) << 4;
        union { uint4 u; uint8_t b[16]; } q;
#pragma unroll
        for (int i = 0; i < 16; ++i) q.b[i] = tile[(mc + i) * 80 + kr];
        *(uint4*)(xqT + (size_t)(kBase + kr) * M + mBase + mc) = q.u;
    }
}

__device__ __forceinline__ v8u tdm_g1_2d(unsigned dim0, unsigned dim1, unsigned stride0) {
    v8u g1;
    g1[0] = (1u << 20) | (4u << 22) | (3u << 25);
    g1[1] = (dim0 & 0xFFFFu) << 16;
    g1[2] = (dim0 >> 16) | ((dim1 & 0xFFFFu) << 16);
    g1[3] = (dim1 >> 16) | (128u << 16);
    g1[4] = 128u;
    g1[5] = stride0;
    g1[6] = 0u;
    g1[7] = 0u;
    return g1;
}

__device__ __forceinline__ void tdm_load_2d(unsigned lds_off, const void* gaddr, v8u g1) {
    unsigned long long ga = (unsigned long long)(uintptr_t)gaddr;
    v4u g0;
    g0[0] = 1u;
    g0[1] = lds_off;
    g0[2] = (unsigned)ga;
    g0[3] = (unsigned)((ga >> 32) & 0x1FFFFFFu) | (2u << 30);
    asm volatile("tensor_load_to_lds %0, %1" :: "s"(g0), "s"(g1) : "memory");
}

__global__ __launch_bounds__(256) void gemm_fp8_kernel(const uint8_t* __restrict__ xqT,
                                                       const uint8_t* __restrict__ w,
                                                       const unsigned* __restrict__ amax_bits,
                                                       const float* __restrict__ w_scale,
                                                       __hip_bfloat16* __restrict__ y,
                                                       int M, int N, int K) {
    extern __shared__ uint8_t smem[];
    const int t = threadIdx.x;
    const int lane = t & 31;
    const int wave = t >> 5;
    const int wN = wave >> 1;
    const int wM = wave & 1;
    const int mBlock = blockIdx.x * 128;
    const int nBlock = blockIdx.y * 128;

    const float outScale = __uint_as_float(*amax_bits) * (1.0f / FP8_MAX_V) * w_scale[0];
    const unsigned ldsBase = (unsigned)(uintptr_t)(void*)smem;

    v8f acc[2][4];
    const v8f vzero = {0.f, 0.f, 0.f, 0.f, 0.f, 0.f, 0.f, 0.f};
#pragma unroll
    for (int i = 0; i < 2; ++i)
#pragma unroll
        for (int j = 0; j < 4; ++j) acc[i][j] = vzero;

    const int laneRow = lane & 15;
    const int kHalfA  = (lane >> 4) << 3;
    const int kLaneB  = (lane & 15) + ((lane >> 4) << 4);

    const v8u g1A = tdm_g1_2d((unsigned)K, (unsigned)N, (unsigned)K);
    const v8u g1B = tdm_g1_2d((unsigned)M, (unsigned)K, (unsigned)M);
    const uint8_t* gA0 = w + (size_t)nBlock * K;
    const uint8_t* gB0 = xqT + mBlock;

    const int KT = K >> 7;

    if (wave == 0) {
        tdm_load_2d(ldsBase, gA0, g1A);
        tdm_load_2d(ldsBase + TILEBYTES, gB0, g1B);
    }

    for (int it = 0; it < KT; ++it) {
        const int cur = it & 1;
        const uint8_t* aT = smem + cur * HALFBYTES;
        const uint8_t* bT = smem + cur * HALFBYTES + TILEBYTES;

        if (wave == 0) {
            if (it + 1 < KT) {
                const int nxt = cur ^ 1;
                const size_t k0n = (size_t)(it + 1) << 7;
                tdm_load_2d(ldsBase + nxt * HALFBYTES,             gA0 + k0n,     g1A);
                tdm_load_2d(ldsBase + nxt * HALFBYTES + TILEBYTES, gB0 + k0n * M, g1B);
                __builtin_amdgcn_s_wait_tensorcnt(2);
            } else {
                __builtin_amdgcn_s_wait_tensorcnt(0);
            }
        }
        __syncthreads();

        v16i afr[2], bfr[4];
#pragma unroll
        for (int ni = 0; ni < 2; ++ni) {
            const uint8_t* base = aT + (wN * 32 + ni * 16 + laneRow) * LDSPITCH + kHalfA;
            union { v16i v; uint2 u[8]; } fa;
#pragma unroll
            for (int p = 0; p < 8; ++p) {
                int kb = ((p >> 2) << 6) + ((p & 3) << 4);
                fa.u[p] = *(const uint2*)(base + kb);
            }
            afr[ni] = fa.v;
        }
#pragma unroll
        for (int mi = 0; mi < 4; ++mi) {
            const uint8_t* base = bT + wM * 64 + mi * 16;
            union { v16i v; uint8_t b[64]; } fb;
#pragma unroll
            for (int q = 0; q < 64; ++q) {
                int kk = ((q >> 5) << 6) + (((q >> 3) & 3) << 4) + kHalfA + (q & 7);
                fb.b[q] = base[kk * LDSPITCH + laneRow];
            }
            bfr[mi] = fb.v;
        }

#pragma unroll
        for (int ni = 0; ni < 2; ++ni)
#pragma unroll
            for (int mi = 0; mi < 4; ++mi)
                acc[ni][mi] = __builtin_amdgcn_wmma_f32_16x16x128_fp8_fp8(
                    afr[ni], bfr[mi], (short)0, acc[ni][mi], false, false);

        __syncthreads();
    }

    const int mOut0 = mBlock + wM * 64 + (lane & 15);
    const int nOut0 = nBlock + wN * 32 + ((lane >> 4) << 3);
#pragma unroll
    for (int ni = 0; ni < 2; ++ni) {
#pragma unroll
        for (int mi = 0; mi < 4; ++mi) {
            union { uint4 u; __hip_bfloat16 h[8]; } pk;
#pragma unroll
            for (int r = 0; r < 8; ++r)
                pk.h[r] = __float2bfloat16(acc[ni][mi][r] * outScale);
            int m = mOut0 + mi * 16;
            int n = nOut0 + ni * 16;
            *(uint4*)(y + (size_t)m * N + n) = pk.u;
        }
    }
}

extern "C" void kernel_launch(void* const* d_in, const int* in_sizes, int n_in,
                              void* d_out, int out_size, void* d_ws, size_t ws_size,
                              hipStream_t stream) {
    const float*   x       = (const float*)d_in[0];
    const uint8_t* w       = (const uint8_t*)d_in[1];
    const float*   w_scale = (const float*)d_in[2];
    __hip_bfloat16* y      = (__hip_bfloat16*)d_out;

    const int K = 4096;
    const int M = in_sizes[0] / K;
    const int N = in_sizes[1] / K;

    unsigned* amax_bits = (unsigned*)d_ws;
    uint8_t*  xqT       = (uint8_t*)d_ws + 256;

    init_amax_kernel<<<1, 1, 0, stream>>>(amax_bits);
    amax_kernel<<<1024, 256, 0, stream>>>(x, amax_bits, (M * K) / 4);

    dim3 qgrid(M / 64, K / 64);
    quantT_kernel<<<qgrid, 256, 0, stream>>>(x, xqT, amax_bits, M, K);

    dim3 ggrid(M / 128, N / 128);
    gemm_fp8_kernel<<<ggrid, 256, 2 * HALFBYTES, stream>>>(xqT, w, amax_bits, w_scale, y, M, N, K);
}
